// GraphSage_network_angles_14087492731320
// MI455X (gfx1250) — matrix operand order repaired and checked by reading, not compiled, not run on hardware
//
#include <hip/hip_runtime.h>

// GraphSage GNN forward for MI455X (gfx1250, wave32, WMMA f16 16x16x32).
// Structure exploits senders = repeat(arange(N), 8): each node owns 8
// contiguous edges, so segment reductions fuse into the edge-MLP kernel
// and happen fully in-lane in the WMMA C layout.

#define NN 100000
#define GG 256

typedef __attribute__((ext_vector_type(16))) _Float16 v16h;
typedef __attribute__((ext_vector_type(8)))  _Float16 v8h;
typedef __attribute__((ext_vector_type(8)))  float    v8f;

__device__ __forceinline__ v8f zero8() {
  v8f z;
#pragma unroll
  for (int i = 0; i < 8; ++i) z[i] = 0.f;
  return z;
}

__device__ __forceinline__ v8f wmma16(v16h a, v16h b, v8f c) {
  // D = A(16x32 f16) * B(32x16 f16) + C(16x16 f32)
  return __builtin_amdgcn_wmma_f32_16x16x32_f16(false, a, false, b, (short)0, c,
                                                false, false);
}

// A-fragment (16x32 f16) from row-major LDS tile. Per ISA layout:
// lane half hh = lane>>4; elements 0..7 -> K = k0+hh*8..+7,
// elements 8..15 -> K = k0+16+hh*8..+7. Two 16B LDS loads per lane.
__device__ __forceinline__ v16h frag_a_lds(const _Float16* base, int lda, int k0) {
  const int lane = threadIdx.x & 31;
  const int row = lane & 15;
  const int hh = (lane >> 4) << 3;
  const _Float16* p = base + row * lda + k0 + hh;
  v8h lo = *(const v8h*)(p);
  v8h hi = *(const v8h*)(p + 16);
  v16h a;
#pragma unroll
  for (int i = 0; i < 8; ++i) { a[i] = lo[i]; a[i + 8] = hi[i]; }
  return a;
}

// B-fragment (32x16 f16) from pre-packed fragment buffer: frag f is
// 32 lanes x 16 halves, contiguous per lane (one 32B load).
__device__ __forceinline__ v16h frag_b_glb(const _Float16* frags, int fidx) {
  const int lane = threadIdx.x & 31;
  return *(const v16h*)(frags + ((size_t)fidx << 9) + (lane << 4));
}

// ---------------- prep kernels ----------------

__global__ void k_norm(const float* __restrict__ x, float* __restrict__ xn, int n) {
  int i = blockIdx.x * blockDim.x + threadIdx.x;
  if (i >= n) return;
  const float T[5] = {0.f, 0.f, -200.f, 10000.f, 0.f};
  const float S[5] = {100.f, 100.f, 100.f, 2500.f, 0.25f};
#pragma unroll
  for (int c = 0; c < 5; ++c)
    xn[(size_t)i * 8 + c] = (x[(size_t)i * 5 + c] - T[c]) / S[c];
  xn[(size_t)i * 8 + 5] = 0.f;
  xn[(size_t)i * 8 + 6] = 0.f;
  xn[(size_t)i * 8 + 7] = 0.f;
}

// Pack row-major f32 W[K][Nf] into f16 B-fragment layout, K padded to 32.
__global__ void k_pack(const float* __restrict__ W, _Float16* __restrict__ dst,
                       int K, int Nf) {
  const int ntiles = Nf >> 4;
  const int ktiles = (K + 31) >> 5;
  const int total = ktiles * ntiles * 512;
  for (int gid = blockIdx.x * blockDim.x + threadIdx.x; gid < total;
       gid += gridDim.x * blockDim.x) {
    int f = gid >> 9, lane = (gid >> 4) & 31, e = gid & 15;
    int kt = f / ntiles, nt = f % ntiles;
    int col = (nt << 4) + (lane & 15);
    int k = (kt << 5) + ((lane >> 4) << 3) + (e & 7) + ((e >> 3) << 4);
    float v = (k < K) ? W[(size_t)k * Nf + col] : 0.f;
    dst[gid] = (_Float16)v;
  }
}

// ---------------- fused edge-MLP + segment agg + node-update MLP ----------------
// Block = 32 nodes = 256 edges. 8 waves; each wave owns two 16-edge tiles.

__global__ __launch_bounds__(256) void k_edge_node(
    const float* __restrict__ xn, const int* __restrict__ recv,
    const float* __restrict__ bng, const float* __restrict__ bnb,
    const float* __restrict__ bnm, const float* __restrict__ bnv,
    const _Float16* __restrict__ m1f, const float* __restrict__ m1b,
    const _Float16* __restrict__ m2f, const float* __restrict__ m2b,
    const _Float16* __restrict__ u1f, const float* __restrict__ u1b,
    const _Float16* __restrict__ u2f, const float* __restrict__ u2b,
    _Float16* __restrict__ h0) {
  __shared__ float s_xn[32][8];
  __shared__ __align__(16) _Float16 emb_s[32][512];   // [node][min|max|mean|var]
  __shared__ __align__(16) union {
    struct { _Float16 a_in[256][32]; _Float16 stage[8][16][64]; } p2;
    struct { _Float16 hid[32][128]; } p3;
  } u;

  const int tid = threadIdx.x;
  const int nbase = blockIdx.x * 32;

  { // sender features into LDS
    int node = tid >> 3, c = tid & 7;
    s_xn[node][c] = xn[(size_t)(nbase + node) * 8 + c];
  }
  __syncthreads();

  { // per-edge 16 input features (K padded to 32 with zeros)
    const int le = tid;
    const int sl = le >> 3;
    const int r = recv[(size_t)nbase * 8 + le];
    float xs[5], xr[5], d[5];
#pragma unroll
    for (int c = 0; c < 5; ++c) {
      xs[c] = s_xn[sl][c];
      xr[c] = xn[(size_t)r * 8 + c];
      d[c] = xr[c] - xs[c];
    }
    float dist = sqrtf(d[0] * d[0] + d[1] * d[1] + d[2] * d[2]);
    float inv = (dist > 0.f) ? (1.f / fmaxf(dist, 1e-30f)) : 0.f;
    float e6[6] = {d[3], d[4], dist, d[0] * inv, d[1] * inv, d[2] * inv};
#pragma unroll
    for (int j = 0; j < 6; ++j)
      e6[j] = (e6[j] - bnm[j]) * rsqrtf(bnv[j] + 1e-3f) * bng[j] + bnb[j];
    _Float16* row = u.p2.a_in[le];
#pragma unroll
    for (int c = 0; c < 5; ++c) { row[c] = (_Float16)xs[c]; row[5 + c] = (_Float16)xr[c]; }
#pragma unroll
    for (int j = 0; j < 6; ++j) row[10 + j] = (_Float16)e6[j];
#pragma unroll
    for (int c = 16; c < 32; ++c) row[c] = (_Float16)0.f;
  }
  __syncthreads();

  const int wave = tid >> 5, lane = tid & 31;
  const int col = lane & 15, rhalf = lane >> 4;

  for (int tt = wave * 2; tt < wave * 2 + 2; ++tt) {
    v8f acc[8];
#pragma unroll
    for (int nt = 0; nt < 8; ++nt) acc[nt] = zero8();

    v16h a1 = frag_a_lds(&u.p2.a_in[tt * 16][0], 32, 0);

    for (int hc = 0; hc < 4; ++hc) {        // hidden(256) in chunks of 64
#pragma unroll
      for (int nt2 = 0; nt2 < 4; ++nt2) {   // layer1: 16 -> 64-chunk
        int nt = hc * 4 + nt2;
        v16h b = frag_b_glb(m1f, nt);
        v8f c = wmma16(a1, b, zero8());
        float bias = m1b[nt * 16 + col];
#pragma unroll
        for (int i = 0; i < 8; ++i) {
          float v = fmaxf(c[i] + bias, 0.f);
          u.p2.stage[wave][rhalf * 8 + i][nt2 * 16 + col] = (_Float16)v;
        }
      }
      asm volatile("s_wait_dscnt 0" ::: "memory");
#pragma unroll
      for (int ks = 0; ks < 2; ++ks) {      // layer2 partial over this chunk
        v16h a2 = frag_a_lds(&u.p2.stage[wave][0][0], 64, ks * 32);
        int kt = hc * 2 + ks;
#pragma unroll
        for (int nt = 0; nt < 8; ++nt) {
          v16h b = frag_b_glb(m2f, kt * 8 + nt);
          acc[nt] = wmma16(a2, b, acc[nt]);
        }
      }
      asm volatile("s_wait_dscnt 0" ::: "memory");
    }

    // bias + relu + in-lane 8-edge min/max/mean/var (C rows 0..7 live in-lane)
    int nl = tt * 2 + rhalf;
#pragma unroll
    for (int nt = 0; nt < 8; ++nt) {
      float bias = m2b[nt * 16 + col];
      float mn = 3.4e38f, mx = -3.4e38f, sm = 0.f, sq = 0.f;
#pragma unroll
      for (int i = 0; i < 8; ++i) {
        float v = fmaxf(acc[nt][i] + bias, 0.f);
        mn = fminf(mn, v); mx = fmaxf(mx, v); sm += v; sq += v * v;
      }
      float mean = sm * 0.125f;
      float var = sq * 0.125f - mean * mean;
      int cc = nt * 16 + col;
      emb_s[nl][cc] = (_Float16)mn;
      emb_s[nl][128 + cc] = (_Float16)mx;
      emb_s[nl][256 + cc] = (_Float16)mean;
      emb_s[nl][384 + cc] = (_Float16)var;
    }
  }
  __syncthreads();

  { // upd1: [32,512] @ [512,128], 8 waves x 2 n-tiles each
    int rt = wave >> 2;
#pragma unroll
    for (int jj = 0; jj < 2; ++jj) {
      int nt = (wave & 3) * 2 + jj;
      v8f acc = zero8();
      for (int kt = 0; kt < 16; ++kt) {
        v16h a = frag_a_lds(&emb_s[rt * 16][0], 512, kt * 32);
        v16h b = frag_b_glb(u1f, kt * 8 + nt);
        acc = wmma16(a, b, acc);
      }
      float bias = u1b[nt * 16 + col];
#pragma unroll
      for (int i = 0; i < 8; ++i) {
        float v = fmaxf(acc[i] + bias, 0.f);
        u.p3.hid[rt * 16 + rhalf * 8 + i][nt * 16 + col] = (_Float16)v;
      }
    }
  }
  __syncthreads();

  { // upd2: [32,128] @ [128,64] -> h0
    int rt = wave >> 2, nt = wave & 3;
    v8f acc = zero8();
#pragma unroll
    for (int kt = 0; kt < 4; ++kt) {
      v16h a = frag_a_lds(&u.p3.hid[rt * 16][0], 128, kt * 32);
      v16h b = frag_b_glb(u2f, kt * 4 + nt);
      acc = wmma16(a, b, acc);
    }
    float bias = u2b[nt * 16 + col];
#pragma unroll
    for (int i = 0; i < 8; ++i) {
      float v = fmaxf(acc[i] + bias, 0.f);
      int node = nbase + rt * 16 + rhalf * 8 + i;
      h0[(size_t)node * 64 + nt * 16 + col] = (_Float16)v;
    }
  }
}

// ---------------- GraphSage conv: h' = relu(l2norm([h, mean_nb(h)] @ W + b)) ----------------

template <int WIN>
__global__ __launch_bounds__(256) void k_gsconv(
    const _Float16* __restrict__ hprev, const int* __restrict__ recv,
    const _Float16* __restrict__ Wf, const float* __restrict__ bias,
    _Float16* __restrict__ hnext) {
  constexpr int WOUT = 2 * WIN;
  constexpr int KT = (2 * WIN) / 32;
  constexpr int NT = WOUT / 16;
  __shared__ __align__(16) _Float16 a_cat[32][2 * WIN];
  __shared__ float o[32][WOUT];
  __shared__ float red[32][8];
  __shared__ float scale[32];

  const int tid = threadIdx.x;
  const int nbase = blockIdx.x * 32;

  { // own features (contiguous 8-half chunks)
    constexpr int CH = WIN / 8;
    for (int i = tid; i < 32 * CH; i += 256) {
      int node = i / CH, c = (i % CH) * 8;
      *(v8h*)&a_cat[node][c] = *(const v8h*)&hprev[(size_t)(nbase + node) * WIN + c];
    }
  }
  { // neighbor mean over the node's 8 contiguous edges
    constexpr int CW = WIN / 8;
    int node = tid >> 3;
    int cs = (tid & 7) * CW;
    float sums[CW];
#pragma unroll
    for (int c = 0; c < CW; ++c) sums[c] = 0.f;
    for (int e = 0; e < 8; ++e) {
      int r = recv[(size_t)(nbase + node) * 8 + e];
      const _Float16* hp = &hprev[(size_t)r * WIN + cs];
#pragma unroll
      for (int c = 0; c < CW; ++c) sums[c] += (float)hp[c];
    }
#pragma unroll
    for (int c = 0; c < CW; ++c)
      a_cat[node][WIN + cs + c] = (_Float16)(sums[c] * 0.125f);
  }
  __syncthreads();

  const int wave = tid >> 5, lane = tid & 31;
  const int col = lane & 15, rhalf = lane >> 4;
  {
    int rt = wave >> 2;
    for (int j = 0; j < NT / 4; ++j) {
      int nt = (wave & 3) + j * 4;
      v8f acc = zero8();
      for (int kt = 0; kt < KT; ++kt) {
        v16h a = frag_a_lds(&a_cat[rt * 16][0], 2 * WIN, kt * 32);
        v16h b = frag_b_glb(Wf, kt * NT + nt);
        acc = wmma16(a, b, acc);
      }
      float bb = bias[nt * 16 + col];
#pragma unroll
      for (int i = 0; i < 8; ++i)
        o[rt * 16 + rhalf * 8 + i][nt * 16 + col] = acc[i] + bb;
    }
  }
  __syncthreads();

  { // l2norm then relu
    constexpr int CWO = WOUT / 8;
    int row = tid >> 3, ch = tid & 7;
    float ssq = 0.f;
#pragma unroll
    for (int c = 0; c < CWO; ++c) { float v = o[row][ch * CWO + c]; ssq += v * v; }
    red[row][ch] = ssq;
    __syncthreads();
    if (tid < 32) {
      float s = 0.f;
#pragma unroll
      for (int j = 0; j < 8; ++j) s += red[tid][j];
      scale[tid] = rsqrtf(fmaxf(s, 1e-12f));
    }
    __syncthreads();
    float sc = scale[row];
#pragma unroll
    for (int c = 0; c < CWO; ++c) {
      float v = fmaxf(o[row][ch * CWO + c] * sc, 0.f);
      hnext[(size_t)(nbase + row) * WOUT + ch * CWO + c] = (_Float16)v;
    }
  }
}

// ---------------- per-graph pooling + decoder + heads ----------------
// graph_idx = (i*G)//N is sorted; graph g owns nodes [ceil(gN/G), ceil((g+1)N/G)).

__global__ __launch_bounds__(256) void k_pool_decode(
    const _Float16* __restrict__ h2,
    const float* W0, const float* b0, const float* g0, const float* be0,
    const float* mu0, const float* va0,
    const float* W1, const float* b1, const float* g1, const float* be1,
    const float* mu1, const float* va1,
    const float* W2, const float* b2, const float* g2, const float* be2,
    const float* mu2, const float* va2,
    const float* A0, const float* a0b, const float* A1, const float* a1b,
    const float* Ao, const float* aob,
    const float* S0, const float* s0b, const float* S1, const float* s1b,
    const float* So, const float* sob,
    float* __restrict__ out) {
  __shared__ float z[768], t0[384], t1[384], t2[192], th[64], th2[64];
  const int g = blockIdx.x, tid = threadIdx.x;
  const int istart = (g * NN + GG - 1) / GG;
  const int iend = ((g + 1) * NN + GG - 1) / GG;
  const float cnt = (float)(iend - istart);
  {
    float mx = -3.4e38f, sm = 0.f;
    for (int n = istart; n < iend; ++n) {
      float v = (float)h2[(size_t)n * 256 + tid];
      mx = fmaxf(mx, v); sm += v;
    }
    z[tid] = mx; z[256 + tid] = sm / cnt; z[512 + tid] = sm;
  }
  __syncthreads();
  for (int o = tid; o < 384; o += 256) {
    float acc = b0[o];
    for (int k = 0; k < 768; ++k) acc += z[k] * W0[k * 384 + o];
    acc = acc > 0.f ? acc : 0.15f * acc;
    t0[o] = (acc - mu0[o]) * rsqrtf(va0[o] + 1e-3f) * g0[o] + be0[o];
  }
  __syncthreads();
  for (int o = tid; o < 384; o += 256) {
    float acc = b1[o];
    for (int k = 0; k < 384; ++k) acc += t0[k] * W1[k * 384 + o];
    acc = acc > 0.f ? acc : 0.15f * acc;
    t1[o] = (acc - mu1[o]) * rsqrtf(va1[o] + 1e-3f) * g1[o] + be1[o];
  }
  __syncthreads();
  for (int o = tid; o < 192; o += 256) {
    float acc = b2[o];
    for (int k = 0; k < 384; ++k) acc += t1[k] * W2[k * 192 + o];
    acc = acc > 0.f ? acc : 0.15f * acc;
    t2[o] = (acc - mu2[o]) * rsqrtf(va2[o] + 1e-3f) * g2[o] + be2[o];
  }
  __syncthreads();
  if (tid < 64) { float a = a0b[tid]; for (int k = 0; k < 192; ++k) a += t2[k] * A0[k * 64 + tid]; th[tid] = a; }
  __syncthreads();
  if (tid < 64) { float a = a1b[tid]; for (int k = 0; k < 64; ++k) a += th[k] * A1[k * 64 + tid]; th2[tid] = a; }
  __syncthreads();
  if (tid < 2) { float a = aob[tid]; for (int k = 0; k < 64; ++k) a += th2[k] * Ao[k * 2 + tid]; out[g * 3 + tid] = a; }
  __syncthreads();
  if (tid < 64) { float a = s0b[tid]; for (int k = 0; k < 192; ++k) a += t2[k] * S0[k * 64 + tid]; th[tid] = a; }
  __syncthreads();
  if (tid < 64) { float a = s1b[tid]; for (int k = 0; k < 64; ++k) a += th[k] * S1[k * 64 + tid]; th2[tid] = a; }
  __syncthreads();
  if (tid == 0) {
    float a = sob[0];
    for (int k = 0; k < 64; ++k) a += th2[k] * So[k];
    out[g * 3 + 2] = fabsf(a) + 1e-5f;
  }
}

// ---------------- launcher ----------------

extern "C" void kernel_launch(void* const* d_in, const int* in_sizes, int n_in,
                              void* d_out, int out_size, void* d_ws, size_t ws_size,
                              hipStream_t stream) {
  (void)in_sizes; (void)n_in; (void)out_size; (void)ws_size;
  const float* x    = (const float*)d_in[0];
  const int*   recv = (const int*)d_in[2];
  const float* bng = (const float*)d_in[4];
  const float* bnb = (const float*)d_in[5];
  const float* bnm = (const float*)d_in[6];
  const float* bnv = (const float*)d_in[7];
  const float* m1W = (const float*)d_in[8];  const float* m1b = (const float*)d_in[9];
  const float* m2W = (const float*)d_in[10]; const float* m2b = (const float*)d_in[11];
  const float* u1W = (const float*)d_in[12]; const float* u1b = (const float*)d_in[13];
  const float* u2W = (const float*)d_in[14]; const float* u2b = (const float*)d_in[15];
  const float* g1W = (const float*)d_in[16]; const float* g1b = (const float*)d_in[17];
  const float* g2W = (const float*)d_in[18]; const float* g2b = (const float*)d_in[19];

  char* ws = (char*)d_ws;
  size_t off = 0;
  auto take = [&](size_t bytes) {
    char* p = ws + off;
    off += (bytes + 255) & ~(size_t)255;
    return p;
  };
  float*    xn   = (float*)take((size_t)NN * 8 * 4);
  _Float16* m1f  = (_Float16*)take(16 * 1024);    // K16(pad32) x 256
  _Float16* m2f  = (_Float16*)take(64 * 1024);    // 256 x 128
  _Float16* u1f  = (_Float16*)take(128 * 1024);   // 512 x 128
  _Float16* u2f  = (_Float16*)take(16 * 1024);    // 128 x 64
  _Float16* gs1f = (_Float16*)take(32 * 1024);    // 128 x 128
  _Float16* gs2f = (_Float16*)take(128 * 1024);   // 256 x 256
  _Float16* h0   = (_Float16*)take((size_t)NN * 64 * 2);
  _Float16* h1   = (_Float16*)take((size_t)NN * 128 * 2);
  _Float16* h2   = (_Float16*)take((size_t)NN * 256 * 2);

  k_norm<<<(NN + 255) / 256, 256, 0, stream>>>(x, xn, NN);
  k_pack<<<256, 256, 0, stream>>>(m1W, m1f, 16, 256);
  k_pack<<<256, 256, 0, stream>>>(m2W, m2f, 256, 128);
  k_pack<<<256, 256, 0, stream>>>(u1W, u1f, 512, 128);
  k_pack<<<256, 256, 0, stream>>>(u2W, u2f, 128, 64);
  k_pack<<<256, 256, 0, stream>>>(g1W, gs1f, 128, 128);
  k_pack<<<256, 256, 0, stream>>>(g2W, gs2f, 256, 256);

  k_edge_node<<<NN / 32, 256, 0, stream>>>(xn, recv, bng, bnb, bnm, bnv,
                                           m1f, m1b, m2f, m2b,
                                           u1f, u1b, u2f, u2b, h0);

  k_gsconv<64><<<NN / 32, 256, 0, stream>>>(h0, recv, gs1f, g1b, h1);
  k_gsconv<128><<<NN / 32, 256, 0, stream>>>(h1, recv, gs2f, g2b, h2);

  k_pool_decode<<<GG, 256, 0, stream>>>(
      h2,
      (const float*)d_in[20], (const float*)d_in[21], (const float*)d_in[22],
      (const float*)d_in[23], (const float*)d_in[24], (const float*)d_in[25],
      (const float*)d_in[26], (const float*)d_in[27], (const float*)d_in[28],
      (const float*)d_in[29], (const float*)d_in[30], (const float*)d_in[31],
      (const float*)d_in[32], (const float*)d_in[33], (const float*)d_in[34],
      (const float*)d_in[35], (const float*)d_in[36], (const float*)d_in[37],
      (const float*)d_in[38], (const float*)d_in[39], (const float*)d_in[40],
      (const float*)d_in[41], (const float*)d_in[42], (const float*)d_in[43],
      (const float*)d_in[44], (const float*)d_in[45], (const float*)d_in[46],
      (const float*)d_in[47], (const float*)d_in[48], (const float*)d_in[49],
      (float*)d_out);
}
